// EdgeConv_33998961115201
// MI455X (gfx1250) — matrix operand order repaired and checked by reading, not compiled, not run on hardware
//
#include <hip/hip_runtime.h>

// ---------------- types ----------------
typedef _Float16 v16h __attribute__((ext_vector_type(16)));
typedef _Float16 h8   __attribute__((ext_vector_type(8)));
typedef float    v8f  __attribute__((ext_vector_type(8)));

// ---------------- fixed problem shape (from reference) ----------------
#define NPT   32768          // N
#define BATCH 4              // B
#define KNB   16             // k
#define OCH   64             // O
#define CCH   64             // C
constexpr float INV_CNT = 1.0f / (4.0f * 32768.0f * 16.0f);   // 1/(B*N*k)
constexpr float BN_EPS  = 1e-5f;

// ---------------- small helpers ----------------
__device__ __forceinline__ v16h cat16(h8 lo, h8 hi) {
  return __builtin_shufflevector(lo, hi, 0,1,2,3,4,5,6,7,8,9,10,11,12,13,14,15);
}

// Load 8 consecutive edge-feature columns [c0, c0+8) for one row.
// Columns 0..63  -> x_i[c];  columns 64..127 -> x_j[c-64] - x_i[c-64].
__device__ __forceinline__ h8 load_e8(const float* __restrict__ xi,
                                      const float* __restrict__ xj, int c0) {
  h8 r;
  if (c0 < 64) {
    float4 u = *(const float4*)(xi + c0);
    float4 v = *(const float4*)(xi + c0 + 4);
    r[0]=(_Float16)u.x; r[1]=(_Float16)u.y; r[2]=(_Float16)u.z; r[3]=(_Float16)u.w;
    r[4]=(_Float16)v.x; r[5]=(_Float16)v.y; r[6]=(_Float16)v.z; r[7]=(_Float16)v.w;
  } else {
    const int cc = c0 - 64;
    float4 uj = *(const float4*)(xj + cc);
    float4 vj = *(const float4*)(xj + cc + 4);
    float4 ui = *(const float4*)(xi + cc);
    float4 vi = *(const float4*)(xi + cc + 4);
    r[0]=(_Float16)(uj.x-ui.x); r[1]=(_Float16)(uj.y-ui.y);
    r[2]=(_Float16)(uj.z-ui.z); r[3]=(_Float16)(uj.w-ui.w);
    r[4]=(_Float16)(vj.x-vi.x); r[5]=(_Float16)(vj.y-vi.y);
    r[6]=(_Float16)(vj.z-vi.z); r[7]=(_Float16)(vj.w-vi.w);
  }
  return r;
}

// B-fragment (K x 16, 16-bit): lane L supplies column o = L%16, K values
// [base, base+8) and [base+16, base+24), base = kb*32 + (L<16 ? 0 : 8).  w points at W[o][0].
__device__ __forceinline__ v16h load_bfrag(const _Float16* w, int base) {
  h8 lo = *(const h8*)(w + base);
  h8 hi = *(const h8*)(w + base + 16);
  return cat16(lo, hi);
}

// ---------------- tiny kernels ----------------
__global__ __launch_bounds__(512) void init_stats_kernel(float* __restrict__ ws) {
  int t = threadIdx.x;
  if (t < 512) ws[t] = 0.0f;
}

// (B,C,N) -> (B,N,C) tiled transpose.  grid (N/32, B*2), block 256.
__global__ __launch_bounds__(256) void transpose_x(const float* __restrict__ x,
                                                   float* __restrict__ xt) {
  __shared__ float tile[32][33];
  const int tx = threadIdx.x & 31;
  const int ty = threadIdx.x >> 5;          // 0..7
  const int n0 = blockIdx.x * 32;
  const int c0 = (blockIdx.y & 1) * 32;
  const int b  = blockIdx.y >> 1;
#pragma unroll
  for (int i = 0; i < 4; ++i) {
    int c = c0 + ty + i * 8;
    tile[ty + i * 8][tx] = x[((size_t)b * CCH + c) * NPT + n0 + tx];
  }
  __syncthreads();
#pragma unroll
  for (int i = 0; i < 4; ++i) {
    int n = n0 + ty + i * 8;
    xt[((size_t)b * NPT + n) * CCH + c0 + tx] = tile[tx][ty + i * 8];
  }
}

// sums[0..63]=sum, sums[64..127]=sumsq  ->  scale/shift for fused BN+affine
__global__ __launch_bounds__(64) void finalize_bn(const float* __restrict__ sums,
                                                  const float* __restrict__ g,
                                                  const float* __restrict__ be,
                                                  float* __restrict__ scale,
                                                  float* __restrict__ shift) {
  int o = threadIdx.x;
  if (o < OCH) {
    float mean = sums[o] * INV_CNT;
    float var  = sums[OCH + o] * INV_CNT - mean * mean;
    float rstd = rsqrtf(var + BN_EPS);
    float sc   = g[o] * rstd;
    scale[o] = sc;
    shift[o] = be[o] - mean * sc;
  }
}

// ---------------- main fused pass ----------------
// PHASE 1: GEMM1 -> per-channel sum/sumsq (layer 1)
// PHASE 2: GEMM1 -> BN1+ReLU -> GEMM2 -> sum/sumsq (layer 2)
// PHASE 3: full pipeline -> max over k -> out (B,O,N)
template <int PHASE>
__global__ __launch_bounds__(256) void edgeconv_pass(
    const float* __restrict__ xt,  const int*   __restrict__ idx,
    const float* __restrict__ W1,  const float* __restrict__ b1,
    const float* __restrict__ W2,  const float* __restrict__ b2,
    const float* __restrict__ sc1, const float* __restrict__ sh1,
    const float* __restrict__ sc2, const float* __restrict__ sh2,
    float* __restrict__ stats,     float* __restrict__ out) {
  __shared__ __align__(16) _Float16 W1s[64][128];   // 16 KB
  __shared__ __align__(16) _Float16 W2s[64][64];    //  8 KB
  __shared__ __align__(16) _Float16 a1s[8][16][72]; // 18 KB, padded rows
  __shared__ float statsS[128];
  __shared__ float mredS[8][64];

  const int tid    = threadIdx.x;
  const int wave   = tid >> 5;
  const int lane   = tid & 31;
  const int lo16   = lane & 15;
  const bool hiH   = lane >= 16;

  for (int i = tid; i < 64 * 128; i += 256)
    ((_Float16*)W1s)[i] = (_Float16)W1[i];
  if (PHASE >= 2)
    for (int i = tid; i < 64 * 64; i += 256)
      ((_Float16*)W2s)[i] = (_Float16)W2[i];
  if (tid < 128) statsS[tid] = 0.0f;
  __syncthreads();

  // one wave == one point (b, n); its 16 neighbors are the 16 WMMA rows
  const int p = blockIdx.x * 8 + wave;            // [0, B*N)
  const int b = p >> 15;                          // N == 2^15
  const int n = p & (NPT - 1);
  const float* xi = xt + (size_t)p * CCH;
  const int    j  = idx[(size_t)p * KNB + lo16];  // lanes L and L+16 share row L%16
  const float* xj = xt + ((size_t)b * NPT + j) * CCH;

  // ---- A fragments for GEMM1: E tile 16 x 128 (f16) ----
  v16h a1f[4];
#pragma unroll
  for (int kb = 0; kb < 4; ++kb) {
    const int base1 = kb * 32 + (hiH ? 8 : 0);
    a1f[kb] = cat16(load_e8(xi, xj, base1), load_e8(xi, xj, base1 + 16));
  }

  // ---- GEMM1: h1(16x64) = E(16x128) x W1^T ----
  v8f acc[4] = {};
#pragma unroll
  for (int kb = 0; kb < 4; ++kb) {
#pragma unroll
    for (int t = 0; t < 4; ++t) {
      v16h bf = load_bfrag(&W1s[t * 16 + lo16][0], kb * 32 + (hiH ? 8 : 0));
      acc[t] = __builtin_amdgcn_wmma_f32_16x16x32_f16(
          false, a1f[kb], false, bf, (short)0, acc[t], false, false);
    }
  }
#pragma unroll
  for (int t = 0; t < 4; ++t) {
    const float bb = b1[t * 16 + lo16];
#pragma unroll
    for (int v = 0; v < 8; ++v) acc[t][v] += bb;
  }

  if (PHASE == 1) {
#pragma unroll
    for (int t = 0; t < 4; ++t) {
      float s = 0.f, q = 0.f;
#pragma unroll
      for (int v = 0; v < 8; ++v) { float h = acc[t][v]; s += h; q += h * h; }
      atomicAdd(&statsS[t * 16 + lo16], s);
      atomicAdd(&statsS[64 + t * 16 + lo16], q);
    }
    __syncthreads();
    if (tid < 128) atomicAdd(&stats[tid], statsS[tid]);
    return;
  }

  // ---- BN1 + ReLU, relayout D-accum -> A-matrix tile in LDS ----
  const int rowoff = hiH ? 8 : 0;
#pragma unroll
  for (int t = 0; t < 4; ++t) {
    const int   o  = t * 16 + lo16;
    const float sc = sc1[o], sh = sh1[o];
#pragma unroll
    for (int v = 0; v < 8; ++v) {
      float h = fmaxf(acc[t][v] * sc + sh, 0.0f);
      a1s[wave][v + rowoff][o] = (_Float16)h;
    }
  }
  __syncthreads();

  // ---- GEMM2: h2(16x64) = A1(16x64) x W2^T ----
  v16h a2f[2];
#pragma unroll
  for (int kb = 0; kb < 2; ++kb) {
    const int base1 = kb * 32 + (hiH ? 8 : 0);
    h8 c1 = *(const h8*)&a1s[wave][lo16][base1];
    h8 c2 = *(const h8*)&a1s[wave][lo16][base1 + 16];
    a2f[kb] = cat16(c1, c2);
  }
  v8f acc2[4] = {};
#pragma unroll
  for (int kb = 0; kb < 2; ++kb) {
#pragma unroll
    for (int t = 0; t < 4; ++t) {
      v16h bf = load_bfrag(&W2s[t * 16 + lo16][0], kb * 32 + (hiH ? 8 : 0));
      acc2[t] = __builtin_amdgcn_wmma_f32_16x16x32_f16(
          false, a2f[kb], false, bf, (short)0, acc2[t], false, false);
    }
  }
#pragma unroll
  for (int t = 0; t < 4; ++t) {
    const float bb = b2[t * 16 + lo16];
#pragma unroll
    for (int v = 0; v < 8; ++v) acc2[t][v] += bb;
  }

  if (PHASE == 2) {
#pragma unroll
    for (int t = 0; t < 4; ++t) {
      float s = 0.f, q = 0.f;
#pragma unroll
      for (int v = 0; v < 8; ++v) { float h = acc2[t][v]; s += h; q += h * h; }
      atomicAdd(&statsS[t * 16 + lo16], s);
      atomicAdd(&statsS[64 + t * 16 + lo16], q);
    }
    __syncthreads();
    if (tid < 128) atomicAdd(&stats[tid], statsS[tid]);
    return;
  }

  // ---- PHASE 3: BN2 + ReLU folded into max over k; out (B,O,N) ----
  float mloc[4];
#pragma unroll
  for (int t = 0; t < 4; ++t) {
    const int   o  = t * 16 + lo16;
    const float sc = sc2[o], sh = sh2[o];
    float m = 0.0f;                         // relu >= 0, so max starts at 0
#pragma unroll
    for (int v = 0; v < 8; ++v) m = fmaxf(m, acc2[t][v] * sc + sh);
    mloc[t] = m;
    if (hiH) mredS[wave][o] = m;            // rows 8..15 half
  }
  __syncthreads();
  if (!hiH) {
#pragma unroll
    for (int t = 0; t < 4; ++t) {
      const int o = t * 16 + lo16;
      float m = fmaxf(mloc[t], mredS[wave][o]);
      out[((size_t)b * OCH + o) * NPT + n] = m;
    }
  }
}

// ---------------- launch ----------------
extern "C" void kernel_launch(void* const* d_in, const int* in_sizes, int n_in,
                              void* d_out, int out_size, void* d_ws, size_t ws_size,
                              hipStream_t stream) {
  (void)in_sizes; (void)n_in; (void)out_size; (void)ws_size;
  const float* x   = (const float*)d_in[0];
  const int*   idx = (const int*)  d_in[1];
  const float* W1  = (const float*)d_in[2];
  const float* b1  = (const float*)d_in[3];
  const float* g1  = (const float*)d_in[4];
  const float* be1 = (const float*)d_in[5];
  const float* W2  = (const float*)d_in[6];
  const float* b2  = (const float*)d_in[7];
  const float* g2  = (const float*)d_in[8];
  const float* be2 = (const float*)d_in[9];
  float* out = (float*)d_out;
  float* ws  = (float*)d_ws;

  float* stats1 = ws;          // 128 floats: sum | sumsq (layer 1)
  float* stats2 = ws + 128;    // 128 floats (layer 2)
  float* sc1 = ws + 256; float* sh1 = ws + 320;
  float* sc2 = ws + 384; float* sh2 = ws + 448;
  float* xt  = ws + 512;       // (B,N,C) transposed x: 32 MB, L2-resident

  init_stats_kernel<<<dim3(1), dim3(512), 0, stream>>>(ws);
  transpose_x<<<dim3(NPT / 32, BATCH * 2), dim3(256), 0, stream>>>(x, xt);

  const dim3 grid(BATCH * NPT / 8);   // 1 wave per point, 8 waves per block
  edgeconv_pass<1><<<grid, 256, 0, stream>>>(xt, idx, W1, b1, W2, b2,
                                             nullptr, nullptr, nullptr, nullptr,
                                             stats1, nullptr);
  finalize_bn<<<dim3(1), dim3(64), 0, stream>>>(stats1, g1, be1, sc1, sh1);
  edgeconv_pass<2><<<grid, 256, 0, stream>>>(xt, idx, W1, b1, W2, b2,
                                             sc1, sh1, nullptr, nullptr,
                                             stats2, nullptr);
  finalize_bn<<<dim3(1), dim3(64), 0, stream>>>(stats2, g2, be2, sc2, sh2);
  edgeconv_pass<3><<<grid, 256, 0, stream>>>(xt, idx, W1, b1, W2, b2,
                                             sc1, sh1, sc2, sh2,
                                             nullptr, out);
}
